// LatticeCharCell_69509750718689
// MI455X (gfx1250) — matrix operand order repaired and checked by reading, not compiled, not run on hardware
//
#include <hip/hip_runtime.h>
#include <hip/hip_bf16.h>

#define B_DIM 4096
#define M_DIM 16
#define H_DIM 512
#define BH    (B_DIM * H_DIM)   // 2097152 elements per [B,H] array
#define WSZ   (512 * 512)       // elements per weight matrix

typedef __attribute__((ext_vector_type(16))) __bf16 v16bf;
typedef __attribute__((ext_vector_type(8)))  __bf16 v8bf;
typedef __attribute__((ext_vector_type(8)))  float  v8f;

__device__ __forceinline__ v8f wmma_bf16(v16bf a, v16bf b, v8f c) {
  return __builtin_amdgcn_wmma_f32_16x16x32_bf16(false, a, false, b, (short)0, c,
                                                 false, false);
}

__device__ __forceinline__ float fast_sigmoid(float x) {
  return 1.0f / (1.0f + __expf(-x));
}
__device__ __forceinline__ float fast_tanh(float x) {
  x = fminf(fmaxf(x, -15.0f), 15.0f);
  float e = __expf(2.0f * x);
  return (e - 1.0f) / (e + 1.0f);
}

// ---- fp32 -> bf16 pre-conversion (8 elements / thread, 16B stores) ---------
__global__ void cvt_bf16_kernel(const float* __restrict__ src,
                                __bf16* __restrict__ dst, int n) {
  int i = (blockIdx.x * blockDim.x + threadIdx.x) * 8;
  if (i >= n) return;
  float4 f0 = ((const float4*)(src + i))[0];
  float4 f1 = ((const float4*)(src + i))[1];
  v8bf o;
  o[0] = (__bf16)f0.x; o[1] = (__bf16)f0.y; o[2] = (__bf16)f0.z; o[3] = (__bf16)f0.w;
  o[4] = (__bf16)f1.x; o[5] = (__bf16)f1.y; o[6] = (__bf16)f1.z; o[7] = (__bf16)f1.w;
  *(v8bf*)(dst + i) = o;
}

// ---- WMMA fragment loaders (bf16 sources, no conversions in hot loops) -----
// A fragment: 16x32 bf16, lane = half*16 + m ; element e:
//   e in [0,7]  -> k = kb + 8*half + e
//   e in [8,15] -> k = kb + 16 + 8*half + (e-8)
__device__ __forceinline__ v16bf load_a_bf(const __bf16* base, int rowBase,
                                           int kb, int lane) {
  int m = lane & 15, half = lane >> 4;
  const __bf16* p = base + (size_t)(rowBase + m) * 512 + kb + half * 8;
  v8bf c0 = *(const v8bf*)p;        // 16B
  v8bf c1 = *(const v8bf*)(p + 16); // 16B
  v16bf a;
#pragma unroll
  for (int i = 0; i < 8; ++i) { a[i] = c0[i]; a[i + 8] = c1[i]; }
  return a;
}

// Same A layout but from bf16 LDS tile [16][512] row-major.
__device__ __forceinline__ v16bf load_a_lds(const __bf16* s, int kb, int lane) {
  int m = lane & 15, half = lane >> 4;
  const __bf16* p = s + m * 512 + kb + half * 8;
  v8bf c0 = *(const v8bf*)p;
  v8bf c1 = *(const v8bf*)(p + 16);
  v16bf a;
#pragma unroll
  for (int i = 0; i < 8; ++i) { a[i] = c0[i]; a[i + 8] = c1[i]; }
  return a;
}

// B fragment: 32x16 bf16, B[k][n] = W[n][k] for x @ W.T.
// lane = half*16 + n ; element e<8 -> k = kb + 8*half + e, e>=8 -> k = kb + 16 + 8*half + (e-8)
// => two aligned 16B loads of 8 contiguous bf16 from row W[hglob].
__device__ __forceinline__ v16bf load_b_bf(const __bf16* W, int hglob, int kb,
                                           int lane) {
  int half = lane >> 4;
  const __bf16* p = W + (size_t)hglob * 512 + kb + half * 8;
  v8bf c0 = *(const v8bf*)p;
  v8bf c1 = *(const v8bf*)(p + 16);
  v16bf b;
#pragma unroll
  for (int i = 0; i < 8; ++i) { b[i] = c0[i]; b[i + 8] = c1[i]; }
  return b;
}

// ---- Kernel 1: gate GEMMs --------------------------------------------------
// One wave per (16-row batch tile, 16-col H tile). Computes
//   i = sigmoid(x@Wii.T + h0@Whi.T + bi), g = tanh(...), o = sigmoid(...),
//   lx = x@Wlx.T + blx   (f gate is dead in the reference -> skipped)
// Stores exp(i), g, o, lx to workspace (fp32 [B,H] each).
__global__ void gates_kernel(const __bf16* __restrict__ Xbf,
                             const __bf16* __restrict__ H0bf,
                             const __bf16* __restrict__ Wbf,  // [ii,hi,ig,hg,io,ho,lx] packed
                             const float* __restrict__ b_ii, const float* __restrict__ b_hi,
                             const float* __restrict__ b_ig, const float* __restrict__ b_hg,
                             const float* __restrict__ b_io, const float* __restrict__ b_ho,
                             const float* __restrict__ b_lx,
                             float* __restrict__ ws) {
  const int lane = threadIdx.x & 31;
  const int wave = threadIdx.x >> 5;
  const int b0 = blockIdx.x * 16;                 // batch rows
  const int h0 = (blockIdx.y * 8 + wave) * 16;    // output columns
  const int n = lane & 15, half = lane >> 4;
  const int hg = h0 + n;

  const __bf16* Wii = Wbf + (size_t)0 * WSZ;
  const __bf16* Whi = Wbf + (size_t)1 * WSZ;
  const __bf16* Wig = Wbf + (size_t)2 * WSZ;
  const __bf16* Whg = Wbf + (size_t)3 * WSZ;
  const __bf16* Wio = Wbf + (size_t)4 * WSZ;
  const __bf16* Who = Wbf + (size_t)5 * WSZ;
  const __bf16* Wlx = Wbf + (size_t)6 * WSZ;

  v8f acc_i = {}, acc_g = {}, acc_o = {}, acc_lx = {};

  for (int kb = 0; kb < 512; kb += 32) {
    v16bf ax = load_a_bf(Xbf, b0, kb, lane);
    v16bf ah = load_a_bf(H0bf, b0, kb, lane);
    v16bf w;
    w = load_b_bf(Wii, hg, kb, lane); acc_i  = wmma_bf16(ax, w, acc_i);
    w = load_b_bf(Whi, hg, kb, lane); acc_i  = wmma_bf16(ah, w, acc_i);
    w = load_b_bf(Wig, hg, kb, lane); acc_g  = wmma_bf16(ax, w, acc_g);
    w = load_b_bf(Whg, hg, kb, lane); acc_g  = wmma_bf16(ah, w, acc_g);
    w = load_b_bf(Wio, hg, kb, lane); acc_o  = wmma_bf16(ax, w, acc_o);
    w = load_b_bf(Who, hg, kb, lane); acc_o  = wmma_bf16(ah, w, acc_o);
    w = load_b_bf(Wlx, hg, kb, lane); acc_lx = wmma_bf16(ax, w, acc_lx);
  }

  const float bi = b_ii[hg] + b_hi[hg];
  const float bg = b_ig[hg] + b_hg[hg];
  const float bo = b_io[hg] + b_ho[hg];
  const float bl = b_lx[hg];

  float* expi = ws;
  float* gw   = ws + (size_t)BH;
  float* ow   = ws + (size_t)2 * BH;
  float* lxw  = ws + (size_t)3 * BH;

#pragma unroll
  for (int r = 0; r < 8; ++r) {
    const int row = b0 + r + 8 * half;        // C/D layout: M = r + 8*half
    const size_t idx = (size_t)row * 512 + hg;
    expi[idx] = __expf(fast_sigmoid(acc_i[r] + bi));
    gw[idx]   = fast_tanh(acc_g[r] + bg);
    ow[idx]   = fast_sigmoid(acc_o[r] + bo);
    lxw[idx]  = acc_lx[r] + bl;
  }
}

// ---- Kernel 2: fused lattice GEMM + masked softmax over words --------------
// One workgroup (8 waves) per batch element b. C_b (16x512) staged to LDS as
// bf16 once; lc tiles live only in WMMA accumulators (never hit memory).
__global__ void lattice_kernel(const float* __restrict__ C,
                               const int* __restrict__ num_word,
                               const __bf16* __restrict__ Wlc,
                               const float* __restrict__ b_lc,
                               const float* __restrict__ ws,
                               float* __restrict__ out) {
  __shared__ __align__(16) __bf16 sC[16 * 512];

  const int b = blockIdx.x;
  const int tid = threadIdx.x;
  const float* Cb = C + (size_t)b * 16 * 512;

  // stage C_b -> LDS bf16 (one-time, off the GEMM critical path)
  for (int idx = tid * 8; idx < 16 * 512; idx += 256 * 8) {
    float4 f0 = ((const float4*)(Cb + idx))[0];
    float4 f1 = ((const float4*)(Cb + idx))[1];
    v8bf o;
    o[0] = (__bf16)f0.x; o[1] = (__bf16)f0.y; o[2] = (__bf16)f0.z; o[3] = (__bf16)f0.w;
    o[4] = (__bf16)f1.x; o[5] = (__bf16)f1.y; o[6] = (__bf16)f1.z; o[7] = (__bf16)f1.w;
    *(v8bf*)(sC + idx) = o;
  }
  __syncthreads();

  const int lane = tid & 31;
  const int wave = tid >> 5;
  const int n = lane & 15, half = lane >> 4;
  const int nw = num_word[b];

  const float* expi = ws;
  const float* gw   = ws + (size_t)BH;
  const float* ow   = ws + (size_t)2 * BH;
  const float* lxw  = ws + (size_t)3 * BH;

  v8f acc[4] = {v8f{}, v8f{}, v8f{}, v8f{}};

  for (int kb = 0; kb < 512; kb += 32) {
    v16bf a = load_a_lds(sC, kb, lane);
#pragma unroll
    for (int j = 0; j < 4; ++j) {
      const int hg = (wave * 4 + j) * 16 + n;
      v16bf w = load_b_bf(Wlc, hg, kb, lane);
      acc[j] = wmma_bf16(a, w, acc[j]);
    }
  }

#pragma unroll
  for (int j = 0; j < 4; ++j) {
    const int hg = (wave * 4 + j) * 16 + n;
    const size_t bh = (size_t)b * 512 + hg;
    const float blc = b_lc[hg];
    const float lxv = lxw[bh];
    const float ei  = expi[bh];

    float sum_e = 0.0f, sum_ec = 0.0f;
#pragma unroll
    for (int r = 0; r < 8; ++r) {
      const int m = r + 8 * half;                       // word index
      const float lc = acc[j][r] + blc;
      const float ib = fast_sigmoid(lxv + lc);
      const float e = (m < nw) ? __expf(ib) : 0.0f;     // masked exp
      sum_e  += e;
      sum_ec += e * Cb[m * 512 + hg];
    }
    // combine the two lane-halves (m 0..7 with m 8..15)
    sum_e  += __shfl_xor(sum_e, 16);
    sum_ec += __shfl_xor(sum_ec, 16);

    const float gv = gw[bh];
    const float ov = ow[bh];
    const float denom = ei + sum_e;
    const float cval = (nw > 0) ? (sum_ec + ei * gv) / denom : gv;
    const float h1 = ov * fast_tanh(cval);

    if (half == 0) {
      out[bh] = h1;                 // h1 output
      out[(size_t)BH + bh] = cval;  // c output
    }
  }
}

extern "C" void kernel_launch(void* const* d_in, const int* in_sizes, int n_in,
                              void* d_out, int out_size, void* d_ws, size_t ws_size,
                              hipStream_t stream) {
  const float* input = (const float*)d_in[0];
  const float* h0    = (const float*)d_in[1];
  // d_in[2] = c0: unused by the reference output
  const float* allC  = (const float*)d_in[3];
  const int*   nword = (const int*)d_in[4];
  const float* W_ii = (const float*)d_in[5],  *b_ii = (const float*)d_in[6];
  const float* W_hi = (const float*)d_in[7],  *b_hi = (const float*)d_in[8];
  // d_in[9..12] = W_if_/b_if_/W_hf/b_hf: f gate is dead in the reference
  const float* W_ig = (const float*)d_in[13], *b_ig = (const float*)d_in[14];
  const float* W_hg = (const float*)d_in[15], *b_hg = (const float*)d_in[16];
  const float* W_io = (const float*)d_in[17], *b_io = (const float*)d_in[18];
  const float* W_ho = (const float*)d_in[19], *b_ho = (const float*)d_in[20];
  const float* W_lx = (const float*)d_in[21], *b_lx = (const float*)d_in[22];
  const float* W_lc = (const float*)d_in[23], *b_lc = (const float*)d_in[24];

  float* ws  = (float*)d_ws;   // layout below; needs ~44 MB
  float* out = (float*)d_out;  // [h1 (B*H) | c (B*H)] fp32

  // ws layout: 4x fp32 [B,H] gate arrays, then bf16 staging areas
  __bf16* Xbf  = (__bf16*)(ws + (size_t)4 * BH);
  __bf16* H0bf = Xbf + (size_t)BH;                 // X/H0 are [B,512]
  __bf16* Wbf  = H0bf + (size_t)BH;                // 8 packed 512x512 weights
  __bf16* Wlcb = Wbf + (size_t)7 * WSZ;

  // --- stage 0: one-time fp32 -> bf16 conversion ---
  const int cvtB = 256;
  const int gW   = WSZ / (cvtB * 8);   // 128 blocks per weight matrix
  const int gX   = BH / (cvtB * 8);    // 1024 blocks for [B,512]
  cvt_bf16_kernel<<<gX, cvtB, 0, stream>>>(input, Xbf, BH);
  cvt_bf16_kernel<<<gX, cvtB, 0, stream>>>(h0, H0bf, BH);
  cvt_bf16_kernel<<<gW, cvtB, 0, stream>>>(W_ii, Wbf + (size_t)0 * WSZ, WSZ);
  cvt_bf16_kernel<<<gW, cvtB, 0, stream>>>(W_hi, Wbf + (size_t)1 * WSZ, WSZ);
  cvt_bf16_kernel<<<gW, cvtB, 0, stream>>>(W_ig, Wbf + (size_t)2 * WSZ, WSZ);
  cvt_bf16_kernel<<<gW, cvtB, 0, stream>>>(W_hg, Wbf + (size_t)3 * WSZ, WSZ);
  cvt_bf16_kernel<<<gW, cvtB, 0, stream>>>(W_io, Wbf + (size_t)4 * WSZ, WSZ);
  cvt_bf16_kernel<<<gW, cvtB, 0, stream>>>(W_ho, Wbf + (size_t)5 * WSZ, WSZ);
  cvt_bf16_kernel<<<gW, cvtB, 0, stream>>>(W_lx, Wbf + (size_t)6 * WSZ, WSZ);
  cvt_bf16_kernel<<<gW, cvtB, 0, stream>>>(W_lc, Wlcb, WSZ);

  // --- stage 1: gate GEMMs (pure bf16 WMMA inner loop) ---
  dim3 g1(B_DIM / 16, (H_DIM / 16) / 8);  // (256, 4), 8 waves/block
  gates_kernel<<<g1, 256, 0, stream>>>(Xbf, H0bf, Wbf,
                                       b_ii, b_hi, b_ig, b_hg,
                                       b_io, b_ho, b_lx, ws);

  // --- stage 2: fused lattice GEMM + masked softmax ---
  lattice_kernel<<<B_DIM, 256, 0, stream>>>(allC, nword, Wlcb, b_lc, ws, out);
}
